// SpectrumPredictor_68745246540012
// MI455X (gfx1250) — compile-verified, matrix operand order checked by reading, not run on hardware
//
#include <hip/hip_runtime.h>
#include <hip/hip_bf16.h>
#include <math.h>

#define P24 24
#define NPAIRS 300
#define NFREQ 512
#define BBATCH 2
#define NPOINTS 4096

typedef _Float16 v16h __attribute__((ext_vector_type(16)));
typedef float    v8f  __attribute__((ext_vector_type(8)));

union Frag { v16h h; unsigned int u[8]; };

// Branch-free erf (Abramowitz & Stegun 7.1.26, |err| <= 1.5e-7) using the
// hardware v_rcp_f32 / v_exp_f32 transcendentals -> straight-line VALU code,
// no exec-mask divergence in the WMMA hot loop (ocml erff is branchy).
__device__ __forceinline__ float fast_erf(float x) {
  float ax = fabsf(x);
  float t = __builtin_amdgcn_rcpf(fmaf(0.3275911f, ax, 1.0f));
  float poly = t * fmaf(t, fmaf(t, fmaf(t, fmaf(t, 1.061405429f, -1.453152027f),
                                        1.421413741f), -0.284496736f), 0.254829592f);
  // exp(-ax^2) = exp2(-ax^2 * log2(e)); underflows cleanly to 0 for large ax
  float e = __builtin_amdgcn_exp2f(ax * ax * -1.4426950408889634f);
  float r = fmaf(-poly, e, 1.0f);
  return copysignf(r, x);
}

__device__ __forceinline__ float gelu_f(float x) {
  return 0.5f * x * (1.0f + fast_erf(x * 0.70710678118654752f));
}

// ---------------------------------------------------------------------------
// 1) Point encoder: 3 -> 64 -> 128 -> 128 (gelu each), partial max per block
//    grid = B*64 blocks, 64 points per block, block = 256 threads
// ---------------------------------------------------------------------------
__global__ __launch_bounds__(256) void point_mlp_kernel(
    const float* __restrict__ points, const float* __restrict__ geom,
    const float* __restrict__ w1, const float* __restrict__ b1,
    const float* __restrict__ w2, const float* __restrict__ b2,
    const float* __restrict__ w3, const float* __restrict__ b3,
    float* __restrict__ partial)
{
  extern __shared__ float S[];
  float* Pt = S;             // 64*3
  float* H1 = S + 192;       // 64*64
  float* H2 = H1 + 64 * 64;  // 64*128
  float* H3 = H2 + 64 * 128; // 64*128
  int b = blockIdx.x >> 6;
  int chunk = blockIdx.x & 63;
  int tid = threadIdx.x;
  float og[3], sc[3];
  for (int d = 0; d < 3; d++) {
    og[d] = geom[b * 6 + d];
    sc[d] = fmaxf(geom[b * 6 + 3 + d], 1e-4f);
  }
  for (int i = tid; i < 64 * 3; i += 256) {
    int pt = i / 3, d = i % 3;
    float v = points[((size_t)(b * NPOINTS) + chunk * 64 + pt) * 3 + d];
    Pt[i] = (v - og[d]) / sc[d];
  }
  __syncthreads();
  for (int i = tid; i < 64 * 64; i += 256) {
    int pt = i >> 6, o = i & 63;
    float acc = b1[o];
    acc += w1[o * 3 + 0] * Pt[pt * 3 + 0] + w1[o * 3 + 1] * Pt[pt * 3 + 1] +
           w1[o * 3 + 2] * Pt[pt * 3 + 2];
    H1[pt * 64 + o] = gelu_f(acc);
  }
  __syncthreads();
  for (int i = tid; i < 64 * 128; i += 256) {
    int pt = i >> 7, o = i & 127;
    float acc = b2[o];
    const float* wr = w2 + o * 64;
    const float* hr = H1 + pt * 64;
    for (int c = 0; c < 64; c++) acc += wr[c] * hr[c];
    H2[pt * 128 + o] = gelu_f(acc);
  }
  __syncthreads();
  for (int i = tid; i < 64 * 128; i += 256) {
    int pt = i >> 7, o = i & 127;
    float acc = b3[o];
    const float* wr = w3 + o * 128;
    const float* hr = H2 + pt * 128;
    for (int c = 0; c < 128; c++) acc += wr[c] * hr[c];
    H3[pt * 128 + o] = gelu_f(acc);
  }
  __syncthreads();
  for (int d = tid; d < 128; d += 256) {
    float m = -3.4e38f;
    for (int pt = 0; pt < 64; pt++) m = fmaxf(m, H3[pt * 128 + d]);
    partial[(size_t)blockIdx.x * 128 + d] = m;
  }
}

// ---------------------------------------------------------------------------
// 2) Global latent: max-reduce partials, concat geom[3:], 131->128 gelu ->128
//    grid = B, block = 128
// ---------------------------------------------------------------------------
__global__ __launch_bounds__(128) void global_latent_kernel(
    const float* __restrict__ partial, const float* __restrict__ geom,
    const float* __restrict__ gw1, const float* __restrict__ gb1,
    const float* __restrict__ gw2, const float* __restrict__ gb2,
    float* __restrict__ glat)
{
  __shared__ float IN[131];
  __shared__ float HB[128];
  int b = blockIdx.x, d = threadIdx.x;
  float m = -3.4e38f;
  for (int c = 0; c < 64; c++) m = fmaxf(m, partial[(size_t)(b * 64 + c) * 128 + d]);
  IN[d] = m;
  if (d < 3) IN[128 + d] = geom[b * 6 + 3 + d];
  __syncthreads();
  float acc = gb1[d];
  const float* wr = gw1 + d * 131;
  for (int c = 0; c < 131; c++) acc += wr[c] * IN[c];
  HB[d] = gelu_f(acc);
  __syncthreads();
  float a2 = gb2[d];
  const float* w2r = gw2 + d * 128;
  for (int c = 0; c < 128; c++) a2 += w2r[c] * HB[c];
  glat[b * 128 + d] = a2;
}

// ---------------------------------------------------------------------------
// 3) Port tokens: pf(16) -> 128 gelu -> 128, + global latent
//    grid = B, block = 128
// ---------------------------------------------------------------------------
__global__ __launch_bounds__(128) void port_tokens_kernel(
    const float* __restrict__ ports, const float* __restrict__ geom,
    const float* __restrict__ glat,
    const float* __restrict__ w1, const float* __restrict__ b1,
    const float* __restrict__ w2, const float* __restrict__ b2,
    float* __restrict__ tokens)
{
  __shared__ float PF[16];
  __shared__ float HB[128];
  int b = blockIdx.x, d = threadIdx.x;
  for (int t = 0; t < P24; t++) {
    if (d == 0) {
      float s[3], e[3], og[3], sc[3];
      for (int k = 0; k < 3; k++) {
        s[k] = ports[(b * P24 + t) * 6 + k];
        e[k] = ports[(b * P24 + t) * 6 + 3 + k];
        og[k] = geom[b * 6 + k];
        sc[k] = fmaxf(geom[b * 6 + 3 + k], 1e-4f);
      }
      float len2 = 0.f;
      for (int k = 0; k < 3; k++) {
        float sl = (s[k] - og[k]) / sc[k];
        float el = (e[k] - og[k]) / sc[k];
        float cl = (0.5f * (s[k] + e[k]) - og[k]) / sc[k];
        float dl = (e[k] - s[k]) / sc[k];
        PF[k] = sl; PF[3 + k] = el; PF[6 + k] = cl; PF[9 + k] = dl;
        PF[13 + k] = sc[k];
        len2 += dl * dl;
      }
      PF[12] = sqrtf(len2);
    }
    __syncthreads();
    float acc = b1[d];
    const float* wr = w1 + d * 16;
    for (int c = 0; c < 16; c++) acc += wr[c] * PF[c];
    HB[d] = gelu_f(acc);
    __syncthreads();
    float a2 = b2[d];
    const float* w2r = w2 + d * 128;
    for (int c = 0; c < 128; c++) a2 += w2r[c] * HB[c];
    tokens[(size_t)(b * P24 + t) * 128 + d] = a2 + glat[b * 128 + d];
    __syncthreads();
  }
}

// ---------------------------------------------------------------------------
// 4) Transformer encoder layer (in-place on tokens), grid = B, block = 256
// ---------------------------------------------------------------------------
__global__ __launch_bounds__(256) void enc_layer_kernel(
    float* __restrict__ tok,
    const float* __restrict__ w_qkv, const float* __restrict__ b_qkv,
    const float* __restrict__ w_o,   const float* __restrict__ b_o,
    const float* __restrict__ w_ff1, const float* __restrict__ b_ff1,
    const float* __restrict__ w_ff2, const float* __restrict__ b_ff2,
    const float* __restrict__ ln1_g, const float* __restrict__ ln1_b,
    const float* __restrict__ ln2_g, const float* __restrict__ ln2_b)
{
  extern __shared__ float S[];
  float* X   = S;                 // 24*128
  float* QKV = S + 3072;          // 24*384, later reused as FFN hidden 24*256
  float* ATT = S + 3072 + 9216;   // 4*24*24
  float* O   = ATT + 2304;        // 24*128
  __shared__ float lnm[24], lnv[24];
  int b = blockIdx.x, tid = threadIdx.x;
  float* tb = tok + (size_t)b * P24 * 128;
  for (int i = tid; i < 3072; i += 256) X[i] = tb[i];
  __syncthreads();
  // QKV projection
  for (int i = tid; i < P24 * 384; i += 256) {
    int t = i / 384, o = i % 384;
    float acc = b_qkv[o];
    const float* wr = w_qkv + o * 128;
    const float* xr = X + t * 128;
    for (int c = 0; c < 128; c++) acc += wr[c] * xr[c];
    QKV[t * 384 + o] = acc;
  }
  __syncthreads();
  // attention scores
  const float scl = 0.17677669529663687f; // 1/sqrt(32)
  for (int i = tid; i < 4 * P24 * P24; i += 256) {
    int h = i / (P24 * P24), r = i % (P24 * P24), q = r / P24, k = r % P24;
    const float* qp = QKV + q * 384 + h * 32;
    const float* kp = QKV + k * 384 + 128 + h * 32;
    float s = 0.f;
    for (int d = 0; d < 32; d++) s += qp[d] * kp[d];
    ATT[i] = s * scl;
  }
  __syncthreads();
  // softmax over 96 rows
  if (tid < 4 * P24) {
    float* row = ATT + tid * P24;
    float m = -3.4e38f;
    for (int k = 0; k < P24; k++) m = fmaxf(m, row[k]);
    float sum = 0.f;
    for (int k = 0; k < P24; k++) { float e = expf(row[k] - m); row[k] = e; sum += e; }
    float inv = 1.f / sum;
    for (int k = 0; k < P24; k++) row[k] *= inv;
  }
  __syncthreads();
  // att @ V
  for (int i = tid; i < 3072; i += 256) {
    int t = i >> 7, d = i & 127, h = d >> 5;
    const float* arow = ATT + (h * P24 + t) * P24;
    float acc = 0.f;
    for (int k = 0; k < P24; k++) acc += arow[k] * QKV[k * 384 + 256 + d];
    O[i] = acc;
  }
  __syncthreads();
  // out-proj + residual
  for (int i = tid; i < 3072; i += 256) {
    int t = i >> 7, d = i & 127;
    float acc = b_o[d];
    const float* wr = w_o + d * 128;
    const float* orow = O + t * 128;
    for (int c = 0; c < 128; c++) acc += wr[c] * orow[c];
    X[i] += acc;
  }
  __syncthreads();
  // LN1
  if (tid < P24) {
    const float* xr = X + tid * 128;
    float m = 0.f;
    for (int c = 0; c < 128; c++) m += xr[c];
    m *= (1.f / 128.f);
    float v = 0.f;
    for (int c = 0; c < 128; c++) { float d2 = xr[c] - m; v += d2 * d2; }
    v *= (1.f / 128.f);
    lnm[tid] = m; lnv[tid] = rsqrtf(v + 1e-5f);
  }
  __syncthreads();
  for (int i = tid; i < 3072; i += 256) {
    int t = i >> 7, d = i & 127;
    X[i] = (X[i] - lnm[t]) * lnv[t] * ln1_g[d] + ln1_b[d];
  }
  __syncthreads();
  // FFN
  float* FH = QKV;
  for (int i = tid; i < P24 * 256; i += 256) {
    int t = i >> 8, o = i & 255;
    float acc = b_ff1[o];
    const float* wr = w_ff1 + o * 128;
    const float* xr = X + t * 128;
    for (int c = 0; c < 128; c++) acc += wr[c] * xr[c];
    FH[t * 256 + o] = gelu_f(acc);
  }
  __syncthreads();
  for (int i = tid; i < 3072; i += 256) {
    int t = i >> 7, d = i & 127;
    float acc = b_ff2[d];
    const float* wr = w_ff2 + d * 256;
    const float* hr = FH + t * 256;
    for (int c = 0; c < 256; c++) acc += wr[c] * hr[c];
    X[i] += acc;
  }
  __syncthreads();
  // LN2
  if (tid < P24) {
    const float* xr = X + tid * 128;
    float m = 0.f;
    for (int c = 0; c < 128; c++) m += xr[c];
    m *= (1.f / 128.f);
    float v = 0.f;
    for (int c = 0; c < 128; c++) { float d2 = xr[c] - m; v += d2 * d2; }
    v *= (1.f / 128.f);
    lnm[tid] = m; lnv[tid] = rsqrtf(v + 1e-5f);
  }
  __syncthreads();
  for (int i = tid; i < 3072; i += 256) {
    int t = i >> 7, d = i & 127;
    tb[i] = (X[i] - lnm[t]) * lnv[t] * ln2_g[d] + ln2_b[d];
  }
}

// ---------------------------------------------------------------------------
// 5) Frequency encoder: fourier feats(17) -> 128 gelu -> 128
//    grid = NFREQ, block = 128
// ---------------------------------------------------------------------------
__global__ __launch_bounds__(128) void freq_latent_kernel(
    const float* __restrict__ fgrid,
    const float* __restrict__ w1, const float* __restrict__ b1,
    const float* __restrict__ w2, const float* __restrict__ b2,
    float* __restrict__ freqlat)
{
  __shared__ float red[256];
  __shared__ float FF[17];
  __shared__ float HB[128];
  int t = threadIdx.x;
  float mn = 3.4e38f, mx = -3.4e38f;
  for (int i = t; i < NFREQ; i += 128) {
    float v = fgrid[i];
    mn = fminf(mn, v); mx = fmaxf(mx, v);
  }
  red[t] = mn; red[128 + t] = mx;
  __syncthreads();
  for (int s = 64; s > 0; s >>= 1) {
    if (t < s) {
      red[t] = fminf(red[t], red[t + s]);
      red[128 + t] = fmaxf(red[128 + t], red[128 + t + s]);
    }
    __syncthreads();
  }
  if (t == 0) {
    float f = (fgrid[blockIdx.x] - red[0]) / (red[128] - red[0] + 1e-6f);
    f = f * 2.0f - 1.0f;
    FF[0] = f;
    for (int i = 0; i < 8; i++) {
      float ang = f * 3.14159265358979323846f * (float)(1 << i);
      FF[1 + i] = sinf(ang);
      FF[9 + i] = cosf(ang);
    }
  }
  __syncthreads();
  float acc = b1[t];
  const float* wr = w1 + t * 17;
  for (int c = 0; c < 17; c++) acc += wr[c] * FF[c];
  HB[t] = gelu_f(acc);
  __syncthreads();
  float a2 = b2[t];
  const float* w2r = w2 + t * 128;
  for (int c = 0; c < 128; c++) a2 += w2r[c] * HB[c];
  freqlat[(size_t)blockIdx.x * 128 + t] = a2;
}

// ---------------------------------------------------------------------------
// 6) Fq[f,o] = dec.b1[o] + sum_c dec.w1[o][128+c] * freq_latent[f][c]
// ---------------------------------------------------------------------------
__global__ __launch_bounds__(256) void dec_fq_kernel(
    const float* __restrict__ freqlat, const float* __restrict__ w1,
    const float* __restrict__ b1, float* __restrict__ Fq)
{
  int f = blockIdx.x, o = threadIdx.x;
  const float* fl = freqlat + (size_t)f * 128;
  const float* wr = w1 + o * 256 + 128;
  float acc = b1[o];
  for (int c = 0; c < 128; c++) acc += wr[c] * fl[c];
  Fq[(size_t)f * 256 + o] = acc;
}

// ---------------------------------------------------------------------------
// 7) Pair encoder: pair_feat(640) -> 256 gelu -> 128, grid = B*NPAIRS
// ---------------------------------------------------------------------------
__global__ __launch_bounds__(256) void pair_kernel(
    const float* __restrict__ tok, const float* __restrict__ glat,
    const float* __restrict__ w1, const float* __restrict__ b1,
    const float* __restrict__ w2, const float* __restrict__ b2,
    float* __restrict__ pairlat)
{
  __shared__ float RT[128], CT[128], GL[128], HB[256];
  int b = blockIdx.x / NPAIRS, p = blockIdx.x % NPAIRS;
  int rr = 0, left = p;
  while (left >= P24 - rr) { left -= P24 - rr; rr++; }
  int cc = rr + left;
  int tid = threadIdx.x;
  for (int i = tid; i < 128; i += 256) {
    RT[i] = tok[(size_t)(b * P24 + rr) * 128 + i];
    CT[i] = tok[(size_t)(b * P24 + cc) * 128 + i];
    GL[i] = glat[b * 128 + i];
  }
  __syncthreads();
  int o = tid;
  float acc = b1[o];
  const float* wr = w1 + (size_t)o * 640;
  for (int j = 0; j < 128; j++) {
    float rt = RT[j], ct = CT[j];
    acc += wr[j] * rt + wr[128 + j] * ct + wr[256 + j] * fabsf(rt - ct) +
           wr[384 + j] * (rt * ct) + wr[512 + j] * GL[j];
  }
  HB[o] = gelu_f(acc);
  __syncthreads();
  if (tid < 128) {
    float a2 = b2[tid];
    const float* w2r = w2 + tid * 256;
    for (int j = 0; j < 256; j++) a2 += w2r[j] * HB[j];
    pairlat[(size_t)(b * NPAIRS + p) * 128 + tid] = a2;
  }
}

// ---------------------------------------------------------------------------
// 8) Apair[bp,o] = sum_c dec.w1[o][c] * pair_latent[bp][c]
// ---------------------------------------------------------------------------
__global__ __launch_bounds__(256) void apair_kernel(
    const float* __restrict__ pairlat, const float* __restrict__ w1,
    float* __restrict__ Apair)
{
  int idx = blockIdx.x;
  int o = threadIdx.x;
  const float* pl = pairlat + (size_t)idx * 128;
  const float* wr = w1 + o * 256;
  float acc = 0.f;
  for (int c = 0; c < 128; c++) acc += wr[c] * pl[c];
  Apair[(size_t)idx * 256 + o] = acc;
}

// ---------------------------------------------------------------------------
// 9) Main decoder: h1 = gelu(A+Fq) (on the fly, f16) ; h2 = gelu(h1 @ W2^T + b2)
//    via v_wmma_f32_16x16x32_f16 ; out = h2 @ W3^T + b3 ; symmetric scatter.
//    grid = 2400 blocks x 256 threads, 128 rows/block (16 rows/wave).
// ---------------------------------------------------------------------------
__global__ __launch_bounds__(256) void dec_main_kernel(
    const float* __restrict__ Apair, const float* __restrict__ Fq,
    const float* __restrict__ w2, const float* __restrict__ b2,
    const float* __restrict__ w3, const float* __restrict__ b3,
    float* __restrict__ out)
{
  extern __shared__ char smem[];
  _Float16* W2s = (_Float16*)smem;                 // 128x256 f16 = 64KB
  _Float16* H1s = (_Float16*)(smem + 65536);       // 8 waves * 16x256 f16 = 64KB
  float* b2s = (float*)(smem + 131072);            // 128
  float* w3s = b2s + 128;                          // 2x128
  float* b3s = w3s + 256;                          // 2

  int tid = threadIdx.x;
  int lane = tid & 31;
  int wave = tid >> 5;
  int r = lane & 15;   // M for A-frag, N for B-frag, column for C/D
  int hi = lane >> 4;

  for (int i = tid; i < 128 * 256; i += 256) W2s[i] = (_Float16)w2[i];
  if (tid < 128) b2s[tid] = b2[tid];
  if (tid < 256) w3s[tid] = w3[tid];
  if (tid < 2) b3s[tid] = b3[tid];

  // stage h1 tile (16 rows x 256) for this wave; row decomposition hoisted
  _Float16* myH = H1s + wave * 16 * 256;
  int r0 = blockIdx.x * 128 + wave * 16;
  for (int m = 0; m < 16; m++) {
    int row = r0 + m;
    int b = row / (NFREQ * NPAIRS);
    int rem = row - b * (NFREQ * NPAIRS);
    int f = rem / NPAIRS;
    int p = rem - f * NPAIRS;
    const float* arow = Apair + ((size_t)(b * NPAIRS + p) << 8);
    const float* frow = Fq + ((size_t)f << 8);
    _Float16* hrow = myH + m * 256;
    for (int c = lane; c < 256; c += 32)
      hrow[c] = (_Float16)gelu_f(arow[c] + frow[c]);
  }
  __syncthreads();

  // fragment k-offsets within a 32-wide K chunk
  int kk[8];
#pragma unroll
  for (int j = 0; j < 8; j++)
    kk[j] = ((j < 4) ? 0 : 16) + hi * 8 + ((j & 3) << 1);

  // load the 8 A fragments (reused across all 8 N tiles)
  Frag af[8];
#pragma unroll
  for (int k = 0; k < 8; k++) {
    const _Float16* base = myH + r * 256 + k * 32;
#pragma unroll
    for (int j = 0; j < 8; j++)
      af[k].u[j] = *(const unsigned int*)(base + kk[j]);
  }

  float o0[8], o1[8];
#pragma unroll
  for (int i = 0; i < 8; i++) { o0[i] = 0.f; o1[i] = 0.f; }

  for (int n = 0; n < 8; n++) {
    v8f acc = {0.f, 0.f, 0.f, 0.f, 0.f, 0.f, 0.f, 0.f};
#pragma unroll
    for (int k = 0; k < 8; k++) {
      Frag bf;
      const _Float16* bb = W2s + (n * 16 + r) * 256 + k * 32;
#pragma unroll
      for (int j = 0; j < 8; j++)
        bf.u[j] = *(const unsigned int*)(bb + kk[j]);
      acc = __builtin_amdgcn_wmma_f32_16x16x32_f16(
          false, af[k].h, false, bf.h, (short)0, acc, false, false);
    }
    int c = n * 16 + r;
    float w3c0 = w3s[c], w3c1 = w3s[128 + c], bb2 = b2s[c];
#pragma unroll
    for (int i = 0; i < 8; i++) {
      float h2 = gelu_f(acc[i] + bb2);
      o0[i] += w3c0 * h2;
      o1[i] += w3c1 * h2;
    }
  }

  // reduce across the 16 lanes of each half-wave (same M set)
#pragma unroll
  for (int i = 0; i < 8; i++) {
    for (int m = 8; m >= 1; m >>= 1) {
      o0[i] += __shfl_xor(o0[i], m, 16);
      o1[i] += __shfl_xor(o1[i], m, 16);
    }
  }

  if (r == 0) {
    for (int i = 0; i < 8; i++) {
      int m = hi * 8 + i;
      int row = r0 + m;
      int b = row / (NFREQ * NPAIRS);
      int rem = row - b * (NFREQ * NPAIRS);
      int f = rem / NPAIRS;
      int p = rem - f * NPAIRS;
      int rr = 0, left = p;
      while (left >= P24 - rr) { left -= P24 - rr; rr++; }
      int cc = rr + left;
      float v0 = o0[i] + b3s[0];
      float v1 = o1[i] + b3s[1];
      size_t base = (size_t)(b * NFREQ + f) * (P24 * P24 * 2);
      out[base + (rr * P24 + cc) * 2 + 0] = v0;
      out[base + (rr * P24 + cc) * 2 + 1] = v1;
      out[base + (cc * P24 + rr) * 2 + 0] = v0;
      out[base + (cc * P24 + rr) * 2 + 1] = v1;
    }
  }
}

// ---------------------------------------------------------------------------
// Host launcher
// ---------------------------------------------------------------------------
extern "C" void kernel_launch(void* const* d_in, const int* in_sizes, int n_in,
                              void* d_out, int out_size, void* d_ws, size_t ws_size,
                              hipStream_t stream) {
  (void)in_sizes; (void)n_in; (void)out_size; (void)ws_size;
  const float* points = (const float*)d_in[0];
  const float* ports  = (const float*)d_in[1];
  const float* geom   = (const float*)d_in[2];
  const float* fgrid  = (const float*)d_in[3];
  // params flattened: top-level insertion order, nested dicts sorted by key
  const float* dec_w1 = (const float*)d_in[4];
  const float* dec_b1 = (const float*)d_in[5];
  const float* dec_w2 = (const float*)d_in[6];
  const float* dec_b2 = (const float*)d_in[7];
  const float* dec_w3 = (const float*)d_in[8];
  const float* dec_b3 = (const float*)d_in[9];
  const float* freq_w1 = (const float*)d_in[10];
  const float* freq_b1 = (const float*)d_in[11];
  const float* freq_w2 = (const float*)d_in[12];
  const float* freq_b2 = (const float*)d_in[13];
  const float* glob_w1 = (const float*)d_in[14];
  const float* glob_b1 = (const float*)d_in[15];
  const float* glob_w2 = (const float*)d_in[16];
  const float* glob_b2 = (const float*)d_in[17];
  const float* pair_w1 = (const float*)d_in[42];
  const float* pair_b1 = (const float*)d_in[43];
  const float* pair_w2 = (const float*)d_in[44];
  const float* pair_b2 = (const float*)d_in[45];
  const float* point_w1 = (const float*)d_in[46];
  const float* point_b1 = (const float*)d_in[47];
  const float* point_w2 = (const float*)d_in[48];
  const float* point_b2 = (const float*)d_in[49];
  const float* point_w3 = (const float*)d_in[50];
  const float* point_b3 = (const float*)d_in[51];
  const float* port_w1 = (const float*)d_in[52];
  const float* port_b1 = (const float*)d_in[53];
  const float* port_w2 = (const float*)d_in[54];
  const float* port_b2 = (const float*)d_in[55];

  float* ws = (float*)d_ws;
  float* partial = ws;             // B*64*128 = 16384
  float* glat    = ws + 16384;     // 256
  float* tok     = ws + 16640;     // B*24*128 = 6144
  float* freqlat = ws + 22784;     // 512*128 = 65536
  float* Fq      = ws + 88320;     // 512*256 = 131072
  float* pairlat = ws + 219392;    // B*300*128 = 76800
  float* Apair   = ws + 296192;    // B*300*256 = 153600

  point_mlp_kernel<<<BBATCH * 64, 256, (192 + 64 * 64 + 64 * 128 * 2) * sizeof(float), stream>>>(
      points, geom, point_w1, point_b1, point_w2, point_b2, point_w3, point_b3, partial);
  global_latent_kernel<<<BBATCH, 128, 0, stream>>>(
      partial, geom, glob_w1, glob_b1, glob_w2, glob_b2, glat);
  port_tokens_kernel<<<BBATCH, 128, 0, stream>>>(
      ports, geom, glat, port_w1, port_b1, port_w2, port_b2, tok);

  for (int l = 0; l < 2; ++l) {
    int LB = 18 + 12 * l;
    const float* b_ff1 = (const float*)d_in[LB + 0];
    const float* b_ff2 = (const float*)d_in[LB + 1];
    const float* b_o   = (const float*)d_in[LB + 2];
    const float* b_qkv = (const float*)d_in[LB + 3];
    const float* ln1_b = (const float*)d_in[LB + 4];
    const float* ln1_g = (const float*)d_in[LB + 5];
    const float* ln2_b = (const float*)d_in[LB + 6];
    const float* ln2_g = (const float*)d_in[LB + 7];
    const float* w_ff1 = (const float*)d_in[LB + 8];
    const float* w_ff2 = (const float*)d_in[LB + 9];
    const float* w_o   = (const float*)d_in[LB + 10];
    const float* w_qkv = (const float*)d_in[LB + 11];
    enc_layer_kernel<<<BBATCH, 256, 17664 * sizeof(float), stream>>>(
        tok, w_qkv, b_qkv, w_o, b_o, w_ff1, b_ff1, w_ff2, b_ff2,
        ln1_g, ln1_b, ln2_g, ln2_b);
  }

  freq_latent_kernel<<<NFREQ, 128, 0, stream>>>(
      fgrid, freq_w1, freq_b1, freq_w2, freq_b2, freqlat);
  dec_fq_kernel<<<NFREQ, 256, 0, stream>>>(freqlat, dec_w1, dec_b1, Fq);
  pair_kernel<<<BBATCH * NPAIRS, 256, 0, stream>>>(
      tok, glat, pair_w1, pair_b1, pair_w2, pair_b2, pairlat);
  apair_kernel<<<BBATCH * NPAIRS, 256, 0, stream>>>(pairlat, dec_w1, Apair);

  const int total_rows = BBATCH * NFREQ * NPAIRS; // 307200
  const int blocks = total_rows / 128;            // 2400
  size_t dec_smem = 65536 + 65536 + (128 + 256 + 2) * sizeof(float);
  dec_main_kernel<<<blocks, 256, dec_smem, stream>>>(
      Apair, Fq, dec_w2, dec_b2, dec_w3, dec_b3, (float*)d_out);
}
